// MultiHeadAttention_12146167513479
// MI455X (gfx1250) — matrix operand order repaired and checked by reading, compiled, not run on hardware
//
#include <hip/hip_runtime.h>
#include <hip/hip_bf16.h>
#include <cstdint>

typedef __attribute__((ext_vector_type(16))) _Float16 v16h;
typedef __attribute__((ext_vector_type(8)))  float    v8f;
typedef __attribute__((ext_vector_type(4)))  int      v4i;

union AFrag { v16h v; uint4 q[2]; };

#define S_LEN  2048
#define NHEAD  16
#define DKV    64
#define DMODEL 1024

#if defined(__has_builtin)
#if __has_builtin(__builtin_amdgcn_global_load_async_to_lds_b128)
#define HAVE_ASYNC_LDS 1
#endif
#if __has_builtin(__builtin_amdgcn_s_wait_asynccnt)
#define HAVE_WAIT_ASYNC 1
#endif
#endif

static __device__ __forceinline__ float hmax16(float x) {
    x = fmaxf(x, __shfl_xor(x, 1, 32));
    x = fmaxf(x, __shfl_xor(x, 2, 32));
    x = fmaxf(x, __shfl_xor(x, 4, 32));
    x = fmaxf(x, __shfl_xor(x, 8, 32));
    return x;
}
static __device__ __forceinline__ float hsum16(float x) {
    x += __shfl_xor(x, 1, 32);
    x += __shfl_xor(x, 2, 32);
    x += __shfl_xor(x, 4, 32);
    x += __shfl_xor(x, 8, 32);
    return x;
}

// ---------------------------------------------------------------------------
// Projection GEMM: out = X @ W + bias, optional RoPE, f16 output in
// split-head layout [B, H, S, DK].
// Block tile 256x64 (8 waves, each 32x64 = 8 WMMAs / BK=32 step).
// ---------------------------------------------------------------------------
#define PBM 256
#define PBN 64
#define PBK 32
#define PKP (PBK + 8)   // padded LDS stride (80 B, 16B-aligned)

__global__ __launch_bounds__(256) void proj_rope_kernel(
    const float* __restrict__ X,     // [M][K]
    const float* __restrict__ W,     // [K][N]
    const float* __restrict__ bias,  // [N]
    _Float16* __restrict__ out,      // [B,H,S,DK]
    int Kdim, int Ndim, int do_rope)
{
    __shared__ __align__(16) _Float16 As[PBM][PKP];   // 20.0 KB
    __shared__ __align__(16) _Float16 Bs[PBN][PKP];   // 5.0 KB, transposed [n][k]

    const int tid   = threadIdx.x;
    const int wave  = tid >> 5;
    const int lane  = tid & 31;
    const int laneN = lane & 15;
    const int half  = (lane >> 4) & 1;
    const int row0  = blockIdx.x * PBM;
    const int col0  = blockIdx.y * PBN;

    v8f acc[2][4] = {};

    for (int k0 = 0; k0 < Kdim; k0 += PBK) {
        __syncthreads();
        {   // A tile: 256x32 fp32 -> f16, 32 contiguous floats / thread
            const float* src = X + (size_t)(row0 + tid) * Kdim + k0;
            #pragma unroll
            for (int j = 0; j < 32; ++j) As[tid][j] = (_Float16)src[j];
        }
        {   // B tile: read W row-major coalesced, store transposed [n][k]
            int kk = tid >> 3, nc = (tid & 7) * 8;
            const float* src = W + (size_t)(k0 + kk) * Ndim + (col0 + nc);
            #pragma unroll
            for (int j = 0; j < 8; ++j) Bs[nc + j][kk] = (_Float16)src[j];
        }
        __syncthreads();

        const int kb = half * 16;
        AFrag a[2];
        #pragma unroll
        for (int s = 0; s < 2; ++s) {
            const int mrow = wave * 32 + s * 16 + laneN;
            a[s].q[0] = *(const uint4*)&As[mrow][kb];
            a[s].q[1] = *(const uint4*)&As[mrow][kb + 8];
        }
        const int kbB = half * 16;
        #pragma unroll
        for (int t = 0; t < 4; ++t) {
            AFrag b;
            const _Float16* bp = &Bs[t * 16 + laneN][0];
            b.q[0] = *(const uint4*)(bp + kbB);
            b.q[1] = *(const uint4*)(bp + kbB + 8);
            #pragma unroll
            for (int s = 0; s < 2; ++s)
                acc[s][t] = __builtin_amdgcn_wmma_f32_16x16x32_f16(
                    false, a[s].v, false, b.v, (short)0, acc[s][t], false, false);
        }
    }

    // Epilogue: bias, RoPE (pairs = adjacent cols = adjacent lanes), f16 store
    const int rowOff = half * 8;
    #pragma unroll
    for (int t = 0; t < 4; ++t) {
        const int n  = col0 + t * 16 + laneN;
        const float bvl = bias[n];
        const int h  = n >> 6;
        const int dk = n & 63;
        #pragma unroll
        for (int s = 0; s < 2; ++s) {
            #pragma unroll
            for (int r = 0; r < 8; ++r) {
                const int gr   = row0 + wave * 32 + s * 16 + r + rowOff;
                const int bidx = gr >> 11;            // / S_LEN
                const int sq   = gr & (S_LEN - 1);
                float val = acc[s][t][r] + bvl;
                float partner = __shfl_xor(val, 1, 32);   // x[2p] <-> x[2p+1]
                if (do_rope) {
                    const float C = -0.28782313662425575f;   // -ln(10000)/32
                    float a_self = (float)sq * __expf((float)(dk & 31) * C);
                    float a_oth  = (float)sq * __expf((float)((dk ^ 1) & 31) * C);
                    float sgn = (dk & 1) ? 1.0f : -1.0f;
                    val = val * __cosf(a_self) + sgn * partner * __sinf(a_oth);
                }
                out[(((size_t)bidx * NHEAD + h) * S_LEN + sq) * DKV + dk] = (_Float16)val;
            }
        }
    }
}

// ---------------------------------------------------------------------------
// Flash attention: 64 queries / block (4 waves x 16 rows), 32-key chunks,
// online softmax, causal mask. Q/K/V f16 [B*H][S][64]; out f16 [B*S][1024].
// K tile arrives via GLOBAL_LOAD_ASYNC_TO_LDS_B128 when available.
// ---------------------------------------------------------------------------
__global__ __launch_bounds__(128) void attn_kernel(
    const _Float16* __restrict__ Q,
    const _Float16* __restrict__ Km,
    const _Float16* __restrict__ Vm,
    _Float16* __restrict__ out)
{
    __shared__ __align__(16) _Float16 Ks[32][72];      // [key][d], padded
    __shared__ __align__(16) _Float16 Vs[64][40];      // [d][key], padded
    __shared__ __align__(16) _Float16 Ps[4][16][40];   // per-wave P patch

    const int tid   = threadIdx.x;
    const int wave  = tid >> 5;
    const int lane  = tid & 31;
    const int laneN = lane & 15;
    const int half  = (lane >> 4) & 1;
    const int rowOff = half * 8;

    const int qblk = blockIdx.x * 64;
    const int bh   = blockIdx.y;          // b*H + h
    const int b    = bh >> 4;
    const int h    = bh & 15;

    const _Float16* Qb = Q  + (size_t)bh * S_LEN * DKV;
    const _Float16* Kb = Km + (size_t)bh * S_LEN * DKV;
    const _Float16* Vb = Vm + (size_t)bh * S_LEN * DKV;

    // Q fragments for d=[0,32) and d=[32,64), contiguous k = 16*half + s
    const int qm = qblk + wave * 16 + laneN;
    const int kb = half * 16;
    AFrag qf[2];
    {
        const _Float16* qr = Qb + (size_t)qm * DKV;
        qf[0].q[0] = *(const uint4*)(qr + kb);
        qf[0].q[1] = *(const uint4*)(qr + kb + 8);
        qf[1].q[0] = *(const uint4*)(qr + 32 + kb);
        qf[1].q[1] = *(const uint4*)(qr + 40 + kb);
    }

    float m[8], l[8];
    #pragma unroll
    for (int r = 0; r < 8; ++r) { m[r] = -1e30f; l[r] = 0.0f; }
    v8f acc[4] = {};

    const int nchunks = qblk / 32 + 2;    // keys [0, qblk+64): causal bound
    for (int kt = 0; kt < nchunks; ++kt) {
        const int key0 = kt * 32;
        __syncthreads();
        {
#if defined(HAVE_ASYNC_LDS)
            // K chunk: 256 x 16B async copies, 2 per thread (ASYNCcnt path)
            #pragma unroll
            for (int cc = 0; cc < 2; ++cc) {
                const int c   = tid + cc * 128;
                const int key = c >> 3;
                const int d8  = (c & 7) * 8;          // halves; 16B chunk
                __builtin_amdgcn_global_load_async_to_lds_b128(
                    (__attribute__((address_space(1))) v4i*)
                        (Kb + (size_t)(key0 + key) * DKV + d8),
                    (__attribute__((address_space(3))) v4i*)&Ks[key][d8],
                    0, 0);
            }
#else
            {
                int key = tid >> 2, d0 = (tid & 3) * 16;
                const _Float16* sk = Kb + (size_t)(key0 + key) * DKV + d0;
                #pragma unroll
                for (int j = 0; j < 16; ++j) Ks[key][d0 + j] = sk[j];
            }
#endif
            // V chunk: transposed store [d][key] (async cannot reshape)
            int key = tid >> 2, d0 = (tid & 3) * 16;
            const _Float16* sv = Vb + (size_t)(key0 + key) * DKV + d0;
            #pragma unroll
            for (int j = 0; j < 16; ++j) Vs[d0 + j][key] = sv[j];
#if defined(HAVE_ASYNC_LDS)
#if defined(HAVE_WAIT_ASYNC)
            __builtin_amdgcn_s_wait_asynccnt(0);
#else
            asm volatile("s_wait_asynccnt 0" ::: "memory");
#endif
#endif
        }
        __syncthreads();

        // scores: two 16x16 tiles, each = 2 WMMAs over d
        v8f sc[2];
        const int kbB = half * 16;
        #pragma unroll
        for (int c = 0; c < 2; ++c) {
            const _Float16* kp = &Ks[c * 16 + laneN][0];
            AFrag b0, b1;
            b0.q[0] = *(const uint4*)(kp + kbB);
            b0.q[1] = *(const uint4*)(kp + kbB + 8);
            b1.q[0] = *(const uint4*)(kp + 32 + kbB);
            b1.q[1] = *(const uint4*)(kp + 32 + kbB + 8);
            v8f s = {};
            s = __builtin_amdgcn_wmma_f32_16x16x32_f16(false, qf[0].v, false, b0.v, (short)0, s, false, false);
            s = __builtin_amdgcn_wmma_f32_16x16x32_f16(false, qf[1].v, false, b1.v, (short)0, s, false, false);
            sc[c] = s;
        }

        // online softmax per row (row state replicated across a 16-lane half)
        #pragma unroll
        for (int r = 0; r < 8; ++r) {
            const int qrow = qblk + wave * 16 + r + rowOff;
            const int k0g = key0 + laneN;
            const int k1g = key0 + 16 + laneN;
            float s0 = (k0g > qrow) ? -1e30f : sc[0][r] * 0.125f;
            float s1 = (k1g > qrow) ? -1e30f : sc[1][r] * 0.125f;
            float mn = fmaxf(m[r], hmax16(fmaxf(s0, s1)));
            float alpha = __expf(m[r] - mn);
            float p0 = __expf(s0 - mn);
            float p1 = __expf(s1 - mn);
            l[r] = l[r] * alpha + hsum16(p0 + p1);
            m[r] = mn;
            #pragma unroll
            for (int t = 0; t < 4; ++t) acc[t][r] *= alpha;
            const int prow = r + rowOff;
            Ps[wave][prow][laneN]      = (_Float16)p0;
            Ps[wave][prow][16 + laneN] = (_Float16)p1;
        }

        // reload P in A-frag layout (same-wave LDS ops are in-order)
        AFrag pf;
        pf.q[0] = *(const uint4*)&Ps[wave][laneN][kb];
        pf.q[1] = *(const uint4*)&Ps[wave][laneN][kb + 8];

        #pragma unroll
        for (int t = 0; t < 4; ++t) {
            const _Float16* vp = &Vs[t * 16 + laneN][0];
            AFrag bv;
            bv.q[0] = *(const uint4*)(vp + kbB);
            bv.q[1] = *(const uint4*)(vp + kbB + 8);
            acc[t] = __builtin_amdgcn_wmma_f32_16x16x32_f16(
                false, pf.v, false, bv.v, (short)0, acc[t], false, false);
        }
    }

    // normalize and write merged-head layout [B*S][D], col = h*64 + d
    #pragma unroll
    for (int r = 0; r < 8; ++r) {
        const float inv = 1.0f / l[r];
        const int qrow = qblk + wave * 16 + r + rowOff;
        const size_t base = ((size_t)b * S_LEN + qrow) * DMODEL + h * DKV;
        #pragma unroll
        for (int t = 0; t < 4; ++t)
            out[base + t * 16 + laneN] = (_Float16)(acc[t][r] * inv);
    }
}

// ---------------------------------------------------------------------------
// Output projection: f16 A @ f32 W + bias -> f32 out (row-major)
// Same 256x64 block tile as proj.
// ---------------------------------------------------------------------------
__global__ __launch_bounds__(256) void oproj_kernel(
    const _Float16* __restrict__ A,   // [M][K] f16
    const float* __restrict__ W,      // [K][N]
    const float* __restrict__ bias,
    float* __restrict__ out,          // [M][N]
    int Kdim, int Ndim)
{
    __shared__ __align__(16) _Float16 As[PBM][PKP];
    __shared__ __align__(16) _Float16 Bs[PBN][PKP];

    const int tid   = threadIdx.x;
    const int wave  = tid >> 5;
    const int lane  = tid & 31;
    const int laneN = lane & 15;
    const int half  = (lane >> 4) & 1;
    const int row0  = blockIdx.x * PBM;
    const int col0  = blockIdx.y * PBN;

    v8f acc[2][4] = {};

    for (int k0 = 0; k0 < Kdim; k0 += PBK) {
        __syncthreads();
        {   // A tile: 256x32 f16, 32 contiguous halves (64 B) / thread
            const _Float16* src = A + (size_t)(row0 + tid) * Kdim + k0;
            #pragma unroll
            for (int j = 0; j < 32; ++j) As[tid][j] = src[j];
        }
        {
            int kk = tid >> 3, nc = (tid & 7) * 8;
            const float* src = W + (size_t)(k0 + kk) * Ndim + (col0 + nc);
            #pragma unroll
            for (int j = 0; j < 8; ++j) Bs[nc + j][kk] = (_Float16)src[j];
        }
        __syncthreads();

        const int kb = half * 16;
        AFrag a[2];
        #pragma unroll
        for (int s = 0; s < 2; ++s) {
            const int mrow = wave * 32 + s * 16 + laneN;
            a[s].q[0] = *(const uint4*)&As[mrow][kb];
            a[s].q[1] = *(const uint4*)&As[mrow][kb + 8];
        }
        const int kbB = half * 16;
        #pragma unroll
        for (int t = 0; t < 4; ++t) {
            AFrag b;
            const _Float16* bp = &Bs[t * 16 + laneN][0];
            b.q[0] = *(const uint4*)(bp + kbB);
            b.q[1] = *(const uint4*)(bp + kbB + 8);
            #pragma unroll
            for (int s = 0; s < 2; ++s)
                acc[s][t] = __builtin_amdgcn_wmma_f32_16x16x32_f16(
                    false, a[s].v, false, b.v, (short)0, acc[s][t], false, false);
        }
    }

    const int rowOff = half * 8;
    #pragma unroll
    for (int t = 0; t < 4; ++t) {
        const int n = col0 + t * 16 + laneN;
        const float bvl = bias[n];
        #pragma unroll
        for (int s = 0; s < 2; ++s) {
            #pragma unroll
            for (int r = 0; r < 8; ++r) {
                const int gr = row0 + wave * 32 + s * 16 + r + rowOff;
                out[(size_t)gr * Ndim + n] = acc[s][t][r] + bvl;
            }
        }
    }
}

// ---------------------------------------------------------------------------
extern "C" void kernel_launch(void* const* d_in, const int* in_sizes, int n_in,
                              void* d_out, int out_size, void* d_ws, size_t ws_size,
                              hipStream_t stream) {
    (void)in_sizes; (void)n_in; (void)out_size; (void)ws_size;
    const float* query = (const float*)d_in[0];
    const float* key   = (const float*)d_in[1];
    const float* value = (const float*)d_in[2];
    const float* Wq = (const float*)d_in[3];
    const float* bq = (const float*)d_in[4];
    const float* Wk = (const float*)d_in[5];
    const float* bk = (const float*)d_in[6];
    const float* Wv = (const float*)d_in[7];
    const float* bv = (const float*)d_in[8];
    const float* Wo = (const float*)d_in[9];
    const float* bo = (const float*)d_in[10];
    float* out = (float*)d_out;

    const int Bn = 2;
    const int M  = Bn * S_LEN;            // 4096 rows
    _Float16* qf   = (_Float16*)d_ws;     // each buffer: 4096*1024 f16 = 8 MB
    _Float16* kf   = qf + (size_t)M * DMODEL;
    _Float16* vf   = kf + (size_t)M * DMODEL;
    _Float16* attn = vf + (size_t)M * DMODEL;

    dim3 pgrid(M / PBM, DMODEL / PBN);    // 16 x 16
    proj_rope_kernel<<<pgrid, 256, 0, stream>>>(query, Wq, bq, qf, DMODEL, DMODEL, 1);
    proj_rope_kernel<<<pgrid, 256, 0, stream>>>(key,   Wk, bk, kf, DMODEL, DMODEL, 1);
    proj_rope_kernel<<<pgrid, 256, 0, stream>>>(value, Wv, bv, vf, DMODEL, DMODEL, 0);

    dim3 agrid(S_LEN / 64, Bn * NHEAD);   // 32 x 32
    attn_kernel<<<agrid, 128, 0, stream>>>(qf, kf, vf, attn);

    oproj_kernel<<<pgrid, 256, 0, stream>>>(attn, Wo, bo, out, DMODEL, DMODEL);
}
